// RNNDecoder_23046794510976
// MI455X (gfx1250) — matrix operand order repaired and checked by reading, not compiled, not run on hardware
//
#include <hip/hip_runtime.h>
#include <hip/hip_bf16.h>

// Problem constants (from the reference)
#define BATCH 64
#define SEQT  256
#define INDIM 512
#define HID   1024
#define G4    (4 * HID)   // 4096 gate columns

typedef __bf16 bf16;
typedef __attribute__((ext_vector_type(16))) __bf16 bf16x16;
typedef __attribute__((ext_vector_type(8)))  __bf16 bf16x8;
typedef __attribute__((ext_vector_type(8)))  float  v8f;

__device__ __forceinline__ bf16 f2bf(float f) {
  // round-to-nearest-even f32 -> bf16
  unsigned u = __builtin_bit_cast(unsigned, f);
  unsigned r = (u + 0x7FFFu + ((u >> 16) & 1u)) >> 16;
  unsigned short s = (unsigned short)r;
  return __builtin_bit_cast(bf16, s);
}

__device__ __forceinline__ float sigmoidf_(float x) {
  return 1.0f / (1.0f + __expf(-x));
}

// ---- WMMA fragment loads -----------------------------------------------
// A: 16x32 bf16 tile of a row-major [rows, ldk] activation matrix.
// Lane layout (ISA 7.12.2): lanes 0-15 -> M=lane, K = kb+{0..7} (elems 0..7)
// and kb+16+{0..7} (elems 8..15); lanes 16-31 -> same M, K bases +8.
__device__ __forceinline__ bf16x16 load_a_frag(const bf16* __restrict__ src,
                                               int ldk, int m0, int kb, int lane) {
  int hi = lane >> 4;
  int m  = m0 + (lane & 15);
  const bf16* p = src + (size_t)m * ldk + kb + hi * 8;
  bf16x16 a;
  ((bf16x8*)&a)[0] = *(const bf16x8*)(p);        // K = base .. base+7
  ((bf16x8*)&a)[1] = *(const bf16x8*)(p + 16);   // K = base+16 .. base+23
  return a;
}

// B: 32x16 bf16 tile of W^T, i.e. 16 rows of row-major W [N_total, ldk].
// Same K layout as A: lanes 0-15 hold K=kb+{0..7},kb+16+{0..7}; lanes 16-31 K bases +8.
__device__ __forceinline__ bf16x16 load_b_frag(const bf16* __restrict__ W,
                                               int ldk, int n0, int kb, int lane) {
  int hi = lane >> 4;
  int n  = n0 + (lane & 15);
  const bf16* p = W + (size_t)n * ldk + kb + hi * 8;
  bf16x16 b;
  ((bf16x8*)&b)[0] = *(const bf16x8*)(p);
  ((bf16x8*)&b)[1] = *(const bf16x8*)(p + 16);
  return b;
}

// ---- Fused LSTM layer step ---------------------------------------------
// gates = x[64,kx] @ Wih^T + h_in[64,H] @ Whh^T + bias ; then cell update.
// One wave per 16x16 (batch x hidden) tile; 4 gate accumulators per wave.
// Grid: 32 blocks x 256 threads = 256 waves = 4 m-tiles * 64 n-tiles.
__global__ __launch_bounds__(256) void lstm_layer_step(
    const bf16* __restrict__ x, int kx,
    const bf16* __restrict__ Wih,     // [4H, kx] row-major bf16
    const bf16* __restrict__ Whh,     // [4H, H]  row-major bf16
    const float* __restrict__ bias,   // [4H] combined b_ih + b_hh
    const bf16* __restrict__ h_in,    // [B, H]
    float* __restrict__ c,            // [B, H] fp32, updated in place
    bf16* __restrict__ h_out)         // [B, H]
{
  int wave = (blockIdx.x * blockDim.x + threadIdx.x) >> 5;
  int lane = threadIdx.x & 31;
  int m_t = wave >> 6;          // 0..3
  int n_t = wave & 63;          // 0..63
  int m0 = m_t * 16;
  int n0 = n_t * 16;
  int n_g = n0 + (lane & 15);   // hidden unit index this lane owns
  int hi = lane >> 4;

  v8f acc[4];
#pragma unroll
  for (int g = 0; g < 4; ++g) {
    float bv = bias[g * HID + n_g];   // bias depends only on N -> broadcast over M
#pragma unroll
    for (int e = 0; e < 8; ++e) acc[g][e] = bv;
  }

  // token/input contribution: x @ Wih^T
  for (int kb = 0; kb < kx; kb += 32) {
    bf16x16 a = load_a_frag(x, kx, m0, kb, lane);
#pragma unroll
    for (int g = 0; g < 4; ++g) {
      bf16x16 b = load_b_frag(Wih + (size_t)g * HID * kx, kx, n0, kb, lane);
      acc[g] = __builtin_amdgcn_wmma_f32_16x16x32_bf16(
          false, a, false, b, (short)0, acc[g], false, false);
    }
  }
  // recurrent contribution: h_in @ Whh^T
  for (int kb = 0; kb < HID; kb += 32) {
    bf16x16 a = load_a_frag(h_in, HID, m0, kb, lane);
#pragma unroll
    for (int g = 0; g < 4; ++g) {
      bf16x16 b = load_b_frag(Whh + (size_t)g * HID * HID, HID, n0, kb, lane);
      acc[g] = __builtin_amdgcn_wmma_f32_16x16x32_bf16(
          false, a, false, b, (short)0, acc[g], false, false);
    }
  }

  // In-register LSTM cell epilogue. D layout: VGPR r, lane -> (M = m0+8*hi+r, N = n_g)
#pragma unroll
  for (int r = 0; r < 8; ++r) {
    int m = m0 + hi * 8 + r;
    size_t idx = (size_t)m * HID + n_g;
    float iv = sigmoidf_(acc[0][r]);
    float fv = sigmoidf_(acc[1][r]);
    float gv = tanhf(acc[2][r]);
    float ov = sigmoidf_(acc[3][r]);
    float cn = fv * c[idx] + iv * gv;
    c[idx] = cn;
    h_out[idx] = f2bf(ov * tanhf(cn));
  }
}

// ---- FC + sigmoid + token feedback -------------------------------------
// out[b,j] = sigmoid(h1[b,:] . fc_w[j,:] + fc_b[j]); writes fp32 output slice
// for step t and the bf16 token for the next step.
// Grid: 16 blocks x 256 threads = 128 waves = 4 m-tiles * 32 n-tiles.
__global__ __launch_bounds__(256) void fc_token_step(
    const bf16* __restrict__ h,       // [B, H]
    const bf16* __restrict__ Wfc,     // [INDIM, H] row-major bf16
    const float* __restrict__ bfc,    // [INDIM]
    bf16* __restrict__ token,         // [B, INDIM] next-step input
    float* __restrict__ out_t)        // d_out + t*INDIM (row stride SEQT*INDIM)
{
  int wave = (blockIdx.x * blockDim.x + threadIdx.x) >> 5;
  int lane = threadIdx.x & 31;
  int m_t = wave >> 5;          // 0..3
  int n_t = wave & 31;          // 0..31
  int m0 = m_t * 16;
  int n0 = n_t * 16;
  int n_g = n0 + (lane & 15);
  int hi = lane >> 4;

  v8f acc;
  {
    float bv = bfc[n_g];
#pragma unroll
    for (int e = 0; e < 8; ++e) acc[e] = bv;
  }
  for (int kb = 0; kb < HID; kb += 32) {
    bf16x16 a = load_a_frag(h, HID, m0, kb, lane);
    bf16x16 b = load_b_frag(Wfc, HID, n0, kb, lane);
    acc = __builtin_amdgcn_wmma_f32_16x16x32_bf16(
        false, a, false, b, (short)0, acc, false, false);
  }
#pragma unroll
  for (int r = 0; r < 8; ++r) {
    int m = m0 + hi * 8 + r;
    float s = sigmoidf_(acc[r]);
    token[(size_t)m * INDIM + n_g] = f2bf(s);
    out_t[(size_t)m * SEQT * INDIM + n_g] = s;
  }
}

// ---- One-time preprocessing kernels ------------------------------------
__global__ void cvt_f32_bf16(const float* __restrict__ s, bf16* __restrict__ d, int n) {
  int i = blockIdx.x * blockDim.x + threadIdx.x;
  if (i < n) d[i] = f2bf(s[i]);
}
__global__ void add_bias_vec(const float* __restrict__ a, const float* __restrict__ b,
                             float* __restrict__ o, int n) {
  int i = blockIdx.x * blockDim.x + threadIdx.x;
  if (i < n) o[i] = a[i] + b[i];
}
__global__ void copy_f32(const float* __restrict__ s, float* __restrict__ d, int n) {
  int i = blockIdx.x * blockDim.x + threadIdx.x;
  if (i < n) d[i] = s[i];
}
__global__ void zero_bf16(bf16* __restrict__ d, int n) {
  int i = blockIdx.x * blockDim.x + threadIdx.x;
  if (i < n) d[i] = f2bf(0.0f);
}

extern "C" void kernel_launch(void* const* d_in, const int* in_sizes, int n_in,
                              void* d_out, int out_size, void* d_ws, size_t ws_size,
                              hipStream_t stream) {
  (void)in_sizes; (void)n_in; (void)out_size; (void)ws_size;
  // setup_inputs order: 0:x(unused) 1:hx 2:c0 3:W_ih0 4:W_hh0 5:b_ih0 6:b_hh0
  //                     7:W_ih1 8:W_hh1 9:b_ih1 10:b_hh1 11:fc_w 12:fc_b
  const float* hx    = (const float*)d_in[1];
  const float* c0in  = (const float*)d_in[2];
  const float* Wih0f = (const float*)d_in[3];
  const float* Whh0f = (const float*)d_in[4];
  const float* bih0  = (const float*)d_in[5];
  const float* bhh0  = (const float*)d_in[6];
  const float* Wih1f = (const float*)d_in[7];
  const float* Whh1f = (const float*)d_in[8];
  const float* bih1  = (const float*)d_in[9];
  const float* bhh1  = (const float*)d_in[10];
  const float* fcwf  = (const float*)d_in[11];
  const float* fcb   = (const float*)d_in[12];
  float* out = (float*)d_out;

  // Workspace layout (bytes); everything 256-aligned by construction.
  char* ws = (char*)d_ws;
  size_t off = 0;
  auto carve = [&](size_t bytes) { char* p = ws + off; off += (bytes + 255) & ~(size_t)255; return p; };
  bf16*  Wih0 = (bf16*)carve((size_t)G4 * INDIM * 2);   // 4MB
  bf16*  Whh0 = (bf16*)carve((size_t)G4 * HID * 2);     // 8MB
  bf16*  Wih1 = (bf16*)carve((size_t)G4 * HID * 2);     // 8MB
  bf16*  Whh1 = (bf16*)carve((size_t)G4 * HID * 2);     // 8MB
  bf16*  Wfc  = (bf16*)carve((size_t)INDIM * HID * 2);  // 1MB
  float* bias0 = (float*)carve((size_t)G4 * 4);
  float* bias1 = (float*)carve((size_t)G4 * 4);
  bf16*  h0buf[2] = {(bf16*)carve((size_t)BATCH * HID * 2), (bf16*)carve((size_t)BATCH * HID * 2)};
  bf16*  h1buf[2] = {(bf16*)carve((size_t)BATCH * HID * 2), (bf16*)carve((size_t)BATCH * HID * 2)};
  float* c0ws = (float*)carve((size_t)BATCH * HID * 4);
  float* c1ws = (float*)carve((size_t)BATCH * HID * 4);
  bf16*  token = (bf16*)carve((size_t)BATCH * INDIM * 2);

  auto blks = [](int n) { return (n + 255) / 256; };

  // --- one-time prep (runs every call; deterministic) ---
  cvt_f32_bf16<<<blks(G4 * INDIM), 256, 0, stream>>>(Wih0f, Wih0, G4 * INDIM);
  cvt_f32_bf16<<<blks(G4 * HID),   256, 0, stream>>>(Whh0f, Whh0, G4 * HID);
  cvt_f32_bf16<<<blks(G4 * HID),   256, 0, stream>>>(Wih1f, Wih1, G4 * HID);
  cvt_f32_bf16<<<blks(G4 * HID),   256, 0, stream>>>(Whh1f, Whh1, G4 * HID);
  cvt_f32_bf16<<<blks(INDIM * HID), 256, 0, stream>>>(fcwf, Wfc, INDIM * HID);
  add_bias_vec<<<blks(G4), 256, 0, stream>>>(bih0, bhh0, bias0, G4);
  add_bias_vec<<<blks(G4), 256, 0, stream>>>(bih1, bhh1, bias1, G4);
  cvt_f32_bf16<<<blks(BATCH * HID), 256, 0, stream>>>(hx,               h0buf[0], BATCH * HID);
  cvt_f32_bf16<<<blks(BATCH * HID), 256, 0, stream>>>(hx + BATCH * HID, h1buf[0], BATCH * HID);
  copy_f32<<<blks(BATCH * HID), 256, 0, stream>>>(c0in,               c0ws, BATCH * HID);
  copy_f32<<<blks(BATCH * HID), 256, 0, stream>>>(c0in + BATCH * HID, c1ws, BATCH * HID);
  zero_bf16<<<blks(BATCH * INDIM), 256, 0, stream>>>(token, BATCH * INDIM);

  // --- 256 sequential decoder steps ---
  int cur = 0;
  for (int t = 0; t < SEQT; ++t) {
    int nxt = cur ^ 1;
    lstm_layer_step<<<32, 256, 0, stream>>>(token, INDIM, Wih0, Whh0, bias0,
                                            h0buf[cur], c0ws, h0buf[nxt]);
    lstm_layer_step<<<32, 256, 0, stream>>>(h0buf[nxt], HID, Wih1, Whh1, bias1,
                                            h1buf[cur], c1ws, h1buf[nxt]);
    fc_token_step<<<16, 256, 0, stream>>>(h1buf[nxt], Wfc, fcb, token,
                                          out + (size_t)t * INDIM);
    cur = nxt;
  }
}
